// Attention_34222299415180
// MI455X (gfx1250) — matrix operand order repaired and checked by reading, not compiled, not run on hardware
//
#include <hip/hip_runtime.h>
#include <hip/hip_bf16.h>
#include <cstdint>
#include <cstddef>

// Problem constants (match reference)
#define B_    4
#define S_    2048
#define D_    2048
#define H_    16
#define KV_   4
#define HD_   128
#define NREP_ (H_ / KV_)

typedef _Float16 f16;
typedef __attribute__((ext_vector_type(16))) _Float16 v16h;
typedef __attribute__((ext_vector_type(8)))  _Float16 v8h;
typedef __attribute__((ext_vector_type(8)))  float    v8f;
typedef __attribute__((ext_vector_type(4)))  float    v4f;

__device__ __forceinline__ v8f wmma32(v16h a, v16h b, v8f c) {
  // D = A(16x32 f16) * B(32x16 f16) + C(16x16 f32)
  return __builtin_amdgcn_wmma_f32_16x16x32_f16(
      /*neg_a=*/false, a, /*neg_b=*/false, b,
      /*c_mod=*/(short)0, c, /*reuse_a=*/false, /*reuse_b=*/false);
}

__device__ __forceinline__ v16h cat8(v8h lo, v8h hi) {
  v16h r;
#pragma unroll
  for (int i = 0; i < 8; ++i) { r[i] = lo[i]; r[i + 8] = hi[i]; }
  return r;
}

// Async copy global -> LDS (GLOBAL_LOAD_ASYNC_TO_LDS_B128, tracked by
// ASYNCcnt). LDS byte offset is the low 32 bits of the generic pointer
__device__ __forceinline__ void async_load_b128(const f16* gptr, f16* lptr) {
  uint32_t lds = (uint32_t)(uintptr_t)lptr;
  uint64_t ga  = (uint64_t)(uintptr_t)gptr;
  asm volatile("global_load_async_to_lds_b128 %0, %1, off"
               :: "v"(lds), "v"(ga)
               : "memory");
}

// ---------------------------------------------------------------------------
// f32 -> f16 elementwise downconvert (vectorized x4)
// ---------------------------------------------------------------------------
__global__ __launch_bounds__(256) void cvt_f32_f16(const float* __restrict__ in,
                                                   f16* __restrict__ out,
                                                   size_t n) {
  size_t i = ((size_t)blockIdx.x * blockDim.x + threadIdx.x) * 4;
  if (i + 4 <= n) {
    v4f v = *(const v4f*)(in + i);
#pragma unroll
    for (int j = 0; j < 4; ++j) out[i + j] = (f16)v[j];
  }
}

// ---------------------------------------------------------------------------
// Generic C[M,N] = A[M,K] * W[N,K]^T, f16 inputs, f32 accum via WMMA.
// Block tile 128x64, 256 threads = 8 waves, each wave computes 32x32
// (2x2 tiles of 16x16). K staged through double-buffered LDS in slabs of
// 32 using GLOBAL_LOAD_ASYNC_TO_LDS_B128: issue slab k+1, compute slab k,
// s_wait_asynccnt 3 (3 in-flight loads per thread for the next slab; async
// loads complete in order so the current slab is guaranteed resident).
//
// A-fragment layout (16-bit A 16x32): lane l (row = l&15) holds
//   K = { (l>>4)*8 .. +7 } and { 16 + (l>>4)*8 .. +7 }  -> two v8h loads.
// B-fragment (32x16, B[k][n] = W[n][k]): lane l (col = l&15) holds
//   K = { (l>>4)*8 .. +7 } and { 16 + (l>>4)*8 .. +7 }  -> two v8h loads.
// ---------------------------------------------------------------------------
template <bool OUT_F32>
__global__ __launch_bounds__(256) void gemm_xwt(const f16* __restrict__ A,
                                                const f16* __restrict__ W,
                                                void* __restrict__ Cout,
                                                int M, int N, int K) {
  __shared__ __attribute__((aligned(64))) f16 As[2][128 * 32];
  __shared__ __attribute__((aligned(64))) f16 Ws[2][64 * 32];

  const int tid  = threadIdx.x;
  const int m0   = blockIdx.y * 128;
  const int n0   = blockIdx.x * 64;
  const int wave = tid >> 5, lane = tid & 31;
  const int l16  = lane & 15, lhi = lane >> 4;
  const int wm   = (wave & 3) * 32;   // 4 waves along M
  const int wn   = (wave >> 2) * 32;  // 2 waves along N

  // Per-thread staging assignment (uniform 3 async b128 per slab):
  //   A tile 128x32 halves: thread -> row tid>>1, k-half (tid&1)*16 (2 loads)
  //   W tile  64x32 halves: thread -> row tid>>2, k-half (tid&3)*8  (1 load)
  const int arow = tid >> 1, akh = (tid & 1) * 16;
  const int wrow = tid >> 2, wkh = (tid & 3) * 8;
  const f16* gA = A + (size_t)(m0 + arow) * K + akh;
  const f16* gW = W + (size_t)(n0 + wrow) * K + wkh;

  auto stage = [&](int k, int buf) {
    async_load_b128(gA + k,     &As[buf][arow * 32 + akh]);
    async_load_b128(gA + k + 8, &As[buf][arow * 32 + akh + 8]);
    async_load_b128(gW + k,     &Ws[buf][wrow * 32 + wkh]);
  };

  v8f acc[2][2] = {};
  const int T = K / 32;
  stage(0, 0);

  for (int it = 0; it < T; ++it) {
    const int cur = it & 1;
    if (it + 1 < T) {
      stage((it + 1) * 32, cur ^ 1);
      if (it + 2 < T)  // global_prefetch_b8, two slabs ahead
        __builtin_prefetch(gA + (it + 2) * 32, 0, 1);
      asm volatile("s_wait_asynccnt 0x3" ::: "memory");
    } else {
      asm volatile("s_wait_asynccnt 0x0" ::: "memory");
    }
    __syncthreads();

#pragma unroll
    for (int mi = 0; mi < 2; ++mi) {
      const f16* ap = &As[cur][(wm + mi * 16 + l16) * 32 + lhi * 8];
      v16h af = cat8(*(const v8h*)ap, *(const v8h*)(ap + 16));
#pragma unroll
      for (int ni = 0; ni < 2; ++ni) {
        const f16* bp = &Ws[cur][(wn + ni * 16 + l16) * 32 + lhi * 8];
        acc[mi][ni] = wmma32(af, cat8(*(const v8h*)bp, *(const v8h*)(bp + 16)), acc[mi][ni]);
      }
    }
    __syncthreads();  // all waves done with buf `cur` before it is re-staged
  }

#pragma unroll
  for (int mi = 0; mi < 2; ++mi)
#pragma unroll
    for (int ni = 0; ni < 2; ++ni) {
      const int col = n0 + wn + ni * 16 + l16;
#pragma unroll
      for (int r = 0; r < 8; ++r) {
        const int row = m0 + wm + mi * 16 + r + lhi * 8;
        if (OUT_F32)
          ((float*)Cout)[(size_t)row * N + col] = acc[mi][ni][r];
        else
          ((f16*)Cout)[(size_t)row * N + col] = (f16)acc[mi][ni][r];
      }
    }
}

// ---------------------------------------------------------------------------
// RoPE (reference layout: pairs (2i,2i+1) -> out[i]=real, out[i+64]=imag)
// ---------------------------------------------------------------------------
__global__ __launch_bounds__(256) void rope_k(const f16* __restrict__ in,
                                              f16* __restrict__ out,
                                              const float* __restrict__ fc,
                                              int nheads, size_t total) {
  size_t idx = (size_t)blockIdx.x * blockDim.x + threadIdx.x;
  if (idx >= total) return;
  const int i = (int)(idx & 63);
  size_t t = idx >> 6;
  const int h = (int)(t % nheads);
  const size_t tok = t / nheads;
  const int s = (int)(tok % S_);
  const size_t base = (tok * nheads + h) * HD_;
  const float x0 = (float)in[base + 2 * i];
  const float x1 = (float)in[base + 2 * i + 1];
  const float c  = fc[((size_t)s * 64 + i) * 2 + 0];
  const float sn = fc[((size_t)s * 64 + i) * 2 + 1];
  out[base + i]      = (f16)(x0 * c - x1 * sn);
  out[base + 64 + i] = (f16)(x0 * sn + x1 * c);
}

// ---------------------------------------------------------------------------
// Flash attention (causal, GQA). Block = 256 threads = 8 waves; each wave
// owns 16 query rows (128 per block). Per 32-key tile:
//   scores: 8x  v_wmma (Q frags resident, K frags streamed from global —
//           K is HD-contiguous so B-fragments are plain b128 loads)
//   online softmax (row stats via __shfl_xor over the 16-lane row groups)
//   P staged via LDS (C-layout -> A-layout reshape), V staged transposed
//   PV: 8x v_wmma into 16x128 f32 accumulators.
// Score matrix never touches HBM.
// ---------------------------------------------------------------------------
__global__ __launch_bounds__(256) void flash_attn(const f16* __restrict__ Q,
                                                  const f16* __restrict__ Kc,
                                                  const f16* __restrict__ V,
                                                  f16* __restrict__ O) {
  __shared__ __attribute__((aligned(64))) f16 Vt[128 * 32];      // [d][key]
  __shared__ __attribute__((aligned(64))) f16 Pls[8 * 16 * 32];  // per-wave P

  const int bh = blockIdx.x;
  const int b = bh / H_, h = bh % H_, g = h / NREP_;
  const int qt = blockIdx.y;
  const int tid = threadIdx.x, wave = tid >> 5, lane = tid & 31;
  const int l16 = lane & 15, lhi = lane >> 4;
  const int q0 = qt * 128 + wave * 16;
  const float scale = 0.08838834764831845f;  // 1/sqrt(128)

  // Q fragments for this wave's 16 rows, all of HD=128 (4 k-steps), resident.
  v16h qf[4];
  {
    const f16* qrow = Q + ((size_t)b * S_ + (q0 + l16)) * (H_ * HD_) + h * HD_;
#pragma unroll
    for (int kk = 0; kk < 4; ++kk) {
      const f16* p = qrow + kk * 32 + lhi * 8;
      qf[kk] = cat8(*(const v8h*)p, *(const v8h*)(p + 16));
    }
  }

  v8f o[8] = {};
  float mrow[8], lrow[8];
#pragma unroll
  for (int r = 0; r < 8; ++r) { mrow[r] = -1e30f; lrow[r] = 0.f; }

  const int ktiles = (qt + 1) * 4;  // causal bound for this 128-row block
  for (int kt = 0; kt < ktiles; ++kt) {
    const int kb = kt * 32;
    __syncthreads();
    {  // stage V tile transposed: Vt[d][key], 32 keys x 128 dims
      const int key = tid >> 3;
      const int db  = (tid & 7) * 16;
      const f16* vp = V + ((size_t)b * S_ + (kb + key)) * (KV_ * HD_) + g * HD_ + db;
      v16h vv = *(const v16h*)vp;
#pragma unroll
      for (int i = 0; i < 16; ++i) Vt[(db + i) * 32 + key] = vv[i];
    }
    __syncthreads();

    // scores: S[16q x 32keys] = Q(16x128) . K(32x128)^T
    v8f s0 = {}, s1 = {};
#pragma unroll
    for (int kk = 0; kk < 4; ++kk) {
      const f16* k0 = Kc + ((size_t)b * S_ + (kb + l16)) * (KV_ * HD_) +
                      g * HD_ + kk * 32 + lhi * 8;
      const f16* k1 = k0 + (size_t)16 * (KV_ * HD_);
      s0 = wmma32(qf[kk], cat8(*(const v8h*)k0, *(const v8h*)(k0 + 16)), s0);
      s1 = wmma32(qf[kk], cat8(*(const v8h*)k1, *(const v8h*)(k1 + 16)), s1);
    }

    // online softmax; C-layout: lane holds col (l&15), rows r + lhi*8
    f16* pw = &Pls[wave * 16 * 32];
#pragma unroll
    for (int r = 0; r < 8; ++r) {
      const int qi = q0 + r + lhi * 8;
      float v0 = s0[r] * scale + ((kb + l16)      <= qi ? 0.f : -1e30f);
      float v1 = s1[r] * scale + ((kb + 16 + l16) <= qi ? 0.f : -1e30f);
      float mx = fmaxf(v0, v1);
#pragma unroll
      for (int off = 1; off < 16; off <<= 1)
        mx = fmaxf(mx, __shfl_xor(mx, off, 16));
      const float mnew  = fmaxf(mrow[r], mx);
      const float alpha = __expf(mrow[r] - mnew);
      const float p0 = __expf(v0 - mnew);
      const float p1 = __expf(v1 - mnew);
      float ps = p0 + p1;
#pragma unroll
      for (int off = 1; off < 16; off <<= 1) ps += __shfl_xor(ps, off, 16);
      lrow[r] = lrow[r] * alpha + ps;
      mrow[r] = mnew;
#pragma unroll
      for (int dt = 0; dt < 8; ++dt) o[dt][r] *= alpha;
      f16* prw = pw + (r + lhi * 8) * 32;
      prw[l16]      = (f16)p0;
      prw[16 + l16] = (f16)p1;
    }
    __syncthreads();

    // P(16x32) . V(32x128): A frag from LDS (now in A-layout addressing)
    v16h pf;
    {
      const f16* pr = pw + l16 * 32 + lhi * 8;
      pf = cat8(*(const v8h*)pr, *(const v8h*)(pr + 16));
    }
#pragma unroll
    for (int dt = 0; dt < 8; ++dt) {
      const f16* vt = &Vt[(dt * 16 + l16) * 32 + lhi * 8];
      o[dt] = wmma32(pf, cat8(*(const v8h*)vt, *(const v8h*)(vt + 16)), o[dt]);
    }
  }

  // normalize and write attention output [B,S,H,HD] as f16
#pragma unroll
  for (int dt = 0; dt < 8; ++dt)
#pragma unroll
    for (int r = 0; r < 8; ++r) {
      const int qi = q0 + r + lhi * 8;
      const int d  = dt * 16 + l16;
      O[((size_t)((size_t)b * S_ + qi) * H_ + h) * HD_ + d] =
          (f16)(o[dt][r] / lrow[r]);
    }
}

// ---------------------------------------------------------------------------
extern "C" void kernel_launch(void* const* d_in, const int* in_sizes, int n_in,
                              void* d_out, int out_size, void* d_ws,
                              size_t ws_size, hipStream_t stream) {
  (void)in_sizes; (void)n_in; (void)out_size; (void)ws_size;
  const float* x  = (const float*)d_in[0];
  const float* fc = (const float*)d_in[1];
  // d_in[2] = dense mask: unused, causal mask applied analytically
  const float* wq = (const float*)d_in[3];
  const float* wk = (const float*)d_in[4];
  const float* wv = (const float*)d_in[5];
  const float* wo = (const float*)d_in[6];
  float* out = (float*)d_out;

  const size_t M = (size_t)B_ * S_;  // 8192 tokens
  uint8_t* wsb = (uint8_t*)d_ws;
  size_t off = 0;
  auto alloc = [&](size_t bytes) { void* p = wsb + off; off += bytes; return p; };
  f16* x16  = (f16*)alloc(M * D_ * 2);
  f16* wq16 = (f16*)alloc((size_t)H_ * HD_ * D_ * 2);
  f16* wk16 = (f16*)alloc((size_t)KV_ * HD_ * D_ * 2);
  f16* wv16 = (f16*)alloc((size_t)KV_ * HD_ * D_ * 2);
  f16* wo16 = (f16*)alloc((size_t)D_ * H_ * HD_ * 2);
  f16* q16  = (f16*)alloc(M * H_ * HD_ * 2);
  f16* k16  = (f16*)alloc(M * KV_ * HD_ * 2);
  f16* v16  = (f16*)alloc(M * KV_ * HD_ * 2);
  f16* qr16 = (f16*)alloc(M * H_ * HD_ * 2);
  f16* kr16 = (f16*)alloc(M * KV_ * HD_ * 2);
  f16* at16 = (f16*)alloc(M * H_ * HD_ * 2);

  auto cvt = [&](const float* in, f16* o16, size_t n) {
    cvt_f32_f16<<<dim3((unsigned)(n / 1024)), dim3(256), 0, stream>>>(in, o16, n);
  };
  cvt(x,  x16,  M * D_);
  cvt(wq, wq16, (size_t)H_ * HD_ * D_);
  cvt(wk, wk16, (size_t)KV_ * HD_ * D_);
  cvt(wv, wv16, (size_t)KV_ * HD_ * D_);
  cvt(wo, wo16, (size_t)D_ * H_ * HD_);

  // QKV projections
  gemm_xwt<false><<<dim3(2048 / 64, (unsigned)(M / 128)), 256, 0, stream>>>(
      x16, wq16, q16, (int)M, 2048, 2048);
  gemm_xwt<false><<<dim3(512 / 64, (unsigned)(M / 128)), 256, 0, stream>>>(
      x16, wk16, k16, (int)M, 512, 2048);
  gemm_xwt<false><<<dim3(512 / 64, (unsigned)(M / 128)), 256, 0, stream>>>(
      x16, wv16, v16, (int)M, 512, 2048);

  // RoPE on Q and K
  {
    const size_t tq = M * H_ * 64;
    rope_k<<<dim3((unsigned)(tq / 256)), 256, 0, stream>>>(q16, qr16, fc, H_, tq);
    const size_t tk = M * KV_ * 64;
    rope_k<<<dim3((unsigned)(tk / 256)), 256, 0, stream>>>(k16, kr16, fc, KV_, tk);
  }

  // Fused causal GQA attention (softmax never leaves the WGP)
  flash_attn<<<dim3(B_ * H_, S_ / 128), 256, 0, stream>>>(qr16, kr16, v16, at16);

  // Output projection, f32 result
  gemm_xwt<true><<<dim3(2048 / 64, (unsigned)(M / 128)), 256, 0, stream>>>(
      at16, wo16, (void*)out, (int)M, 2048, 2048);
}
